// LinearPolicy_82746839924987
// MI455X (gfx1250) — matrix operand order repaired and checked by reading, not compiled, not run on hardware
//
#include <hip/hip_runtime.h>
#include <hip/hip_bf16.h>
#include <math.h>
#include <stdint.h>

typedef __attribute__((ext_vector_type(16))) __bf16 v16bf;
typedef __attribute__((ext_vector_type(8)))  __bf16 v8bf;
typedef __attribute__((ext_vector_type(8)))  float  v8f;

#define ACT_NONE 0
#define ACT_GELU 1
#define ACT_CLIP 2

// Flip to 0 if the assembler rejects the async mnemonic.
#ifndef USE_ASYNC_COPY
#define USE_ASYNC_COPY 1
#endif

__device__ __forceinline__ unsigned short f2bf(float f) {
    unsigned int u = __float_as_uint(f);
    unsigned int r = u + 0x7FFFu + ((u >> 16) & 1u);   // RNE f32 -> bf16
    return (unsigned short)(r >> 16);
}
__device__ __forceinline__ float bf2f(unsigned short u) {
    return __uint_as_float(((unsigned int)u) << 16);
}

// 16-byte global(bf16) -> LDS copy. Async path uses the CDNA5 ASYNCcnt DMA
// (global_load_async_to_lds_b128: LDS-addr VGPR, 64-bit addr VGPR pair).
// Flat shared-aperture pointers carry the LDS byte address in the low 32 bits.
__device__ __forceinline__ void copy16_to_lds(unsigned short* ldst,
                                              const unsigned short* gsrc) {
#if USE_ASYNC_COPY
    unsigned int loff = (unsigned int)(uintptr_t)ldst;
    asm volatile("global_load_async_to_lds_b128 %0, %1, off"
                 :
                 : "v"(loff), "v"((unsigned long long)(uintptr_t)gsrc)
                 : "memory");
#else
    *(uint4*)ldst = *(const uint4*)gsrc;
#endif
}
__device__ __forceinline__ void wait_lds_copies() {
#if USE_ASYNC_COPY
#if __has_builtin(__builtin_amdgcn_s_wait_asynccnt)
    __builtin_amdgcn_s_wait_asynccnt(0);
#else
    asm volatile("s_wait_asynccnt 0x0" ::: "memory");
#endif
#endif
}

// ---------------------------------------------------------------------------
// bf16 WMMA GEMM: out[M,N] = act(A[M,K] @ Wt[N,K]^T + bias (+ residual))
// A, Wt already bf16 (Wt stored transposed: row n = output column n).
// Block 256 threads (8 waves); block tile 64(M) x 128(N); wave tile 32x32.
// ---------------------------------------------------------------------------
template<int ACT, bool RES, bool OUTBF>
__global__ __launch_bounds__(256) void gemm_wmma_kernel(
    const unsigned short* __restrict__ Abf, const unsigned short* __restrict__ Wt,
    const float* __restrict__ bias, const float* __restrict__ res,
    int resMod, void* __restrict__ outv, int M, int N, int K)
{
    __shared__ unsigned short As[64 * 32];    // [row][k]  4 KB
    __shared__ unsigned short Bs[128 * 32];   // [n][k]    8 KB

    const int tid   = threadIdx.x;
    const int lane  = tid & 31;
    const int wid   = tid >> 5;
    const int warpM = wid >> 2;               // 0..1
    const int warpN = wid & 3;                // 0..3
    const int blockM = blockIdx.y * 64;
    const int blockN = blockIdx.x * 128;
    const int lr = lane & 15;
    const int lh = lane >> 4;

    // staging coordinates
    const int arow = tid >> 2, aseg = tid & 3;        // A: 64 x (4 x 8 bf16)
    const int bn   = tid >> 1, bhalf = tid & 1;       // B: 128 x (2 x 16 bf16)
    const int gnb  = blockN + bn;
    const int bsrow = (gnb < N) ? gnb : (N - 1);      // clamp (cols discarded)

    v8f acc[2][2];
#pragma unroll
    for (int mi = 0; mi < 2; ++mi)
#pragma unroll
        for (int ni = 0; ni < 2; ++ni)
#pragma unroll
            for (int j = 0; j < 8; ++j) acc[mi][ni][j] = 0.0f;

    for (int k0 = 0; k0 < K; k0 += 32) {
        // ---- stage A tile (64x32 bf16) ----
        copy16_to_lds(&As[arow * 32 + aseg * 8],
                      Abf + (size_t)(blockM + arow) * K + k0 + aseg * 8);
        // ---- stage B tile (128x32 bf16, already transposed in memory) ----
        {
            const unsigned short* bsrc = Wt + (size_t)bsrow * K + k0 + bhalf * 16;
            unsigned short* bdst = &Bs[bn * 32 + bhalf * 16];
            copy16_to_lds(bdst, bsrc);
            copy16_to_lds(bdst + 8, bsrc + 8);
            if (k0 + 32 < K)  // -> global_prefetch_b8
                __builtin_prefetch(bsrc + 32, 0, 1);
        }
        wait_lds_copies();
        __syncthreads();

        // ---- fragments per ISA lane layouts ----
        v16bf afrag[2], bfrag[2];
#pragma unroll
        for (int mi = 0; mi < 2; ++mi) {
            const unsigned short* base = &As[(warpM * 32 + mi * 16 + lr) * 32];
            v8bf lo = *(const v8bf*)(base + lh * 8);
            v8bf hi = *(const v8bf*)(base + 16 + lh * 8);
#pragma unroll
            for (int i = 0; i < 8; ++i) { afrag[mi][i] = lo[i]; afrag[mi][i + 8] = hi[i]; }
        }
#pragma unroll
        for (int ni = 0; ni < 2; ++ni) {
            const unsigned short* base = &Bs[(warpN * 32 + ni * 16 + lr) * 32];
            v8bf lo = *(const v8bf*)(base + lh * 8);
            v8bf hi = *(const v8bf*)(base + 16 + lh * 8);
#pragma unroll
            for (int i = 0; i < 8; ++i) { bfrag[ni][i] = lo[i]; bfrag[ni][i + 8] = hi[i]; }
        }

#pragma unroll
        for (int mi = 0; mi < 2; ++mi)
#pragma unroll
            for (int ni = 0; ni < 2; ++ni)
                acc[mi][ni] = __builtin_amdgcn_wmma_f32_16x16x32_bf16(
                    false, afrag[mi], false, bfrag[ni],
                    (short)0, acc[mi][ni], false, false);
        __syncthreads();
    }

    // ---- epilogue ----
#pragma unroll
    for (int mi = 0; mi < 2; ++mi) {
#pragma unroll
        for (int ni = 0; ni < 2; ++ni) {
            const int gn = blockN + warpN * 32 + ni * 16 + lr;
            if (gn >= N) continue;
            const float bv = bias[gn];
#pragma unroll
            for (int j = 0; j < 8; ++j) {
                const int gm = blockM + warpM * 32 + mi * 16 + lh * 8 + j;
                float v = acc[mi][ni][j] + bv;
                if (RES) {
                    const int rr = resMod ? (gm % resMod) : gm;
                    v += res[(size_t)rr * N + gn];
                }
                if (ACT == ACT_GELU) v = 0.5f * v * (1.0f + erff(v * 0.70710678118654752f));
                if (ACT == ACT_CLIP) v = fminf(fmaxf(v, -2.0f), 1.0f);
                if (OUTBF) ((unsigned short*)outv)[(size_t)gm * N + gn] = f2bf(v);
                else       ((float*)outv)[(size_t)gm * N + gn] = v;
            }
        }
    }
}

// ---------------------------------------------------------------------------
// Transpose + convert: src[K,N] fp32  ->  dst[N,K] bf16
// ---------------------------------------------------------------------------
__global__ __launch_bounds__(256) void transpose_bf16_kernel(
    const float* __restrict__ src, unsigned short* __restrict__ dst, int K, int N)
{
    __shared__ float tile[32][33];
    const int tx = threadIdx.x & 31;
    const int ty = threadIdx.x >> 5;          // 0..7
    const int kb = blockIdx.y * 32;
    const int nb = blockIdx.x * 32;
#pragma unroll
    for (int i = 0; i < 4; ++i) {
        const int k = kb + ty + i * 8, n = nb + tx;
        tile[ty + i * 8][tx] = (k < K && n < N) ? src[(size_t)k * N + n] : 0.0f;
    }
    __syncthreads();
#pragma unroll
    for (int i = 0; i < 4; ++i) {
        const int n = nb + ty + i * 8, k = kb + tx;
        if (n < N && k < K) dst[(size_t)n * K + k] = f2bf(tile[tx][ty + i * 8]);
    }
}

// elementwise fp32 -> bf16
__global__ __launch_bounds__(256) void convert_bf16_kernel(
    const float* __restrict__ src, unsigned short* __restrict__ dst, int n)
{
    for (int i = blockIdx.x * blockDim.x + threadIdx.x; i < n; i += gridDim.x * blockDim.x)
        dst[i] = f2bf(src[i]);
}

// ---------------------------------------------------------------------------
// LayerNorm (C=512), fp32 in -> bf16 out. One 256-thread block per row.
// ---------------------------------------------------------------------------
__global__ __launch_bounds__(256) void ln_kernel(
    const float* __restrict__ x, const float* __restrict__ g,
    const float* __restrict__ b, unsigned short* __restrict__ y, int C)
{
    const int row = blockIdx.x;
    const int tid = threadIdx.x;
    const float* xr = x + (size_t)row * C;
    const float v0 = xr[tid];
    const float v1 = xr[tid + 256];
    __shared__ float s1[256], s2[256];
    s1[tid] = v0 + v1;
    s2[tid] = v0 * v0 + v1 * v1;
    __syncthreads();
    for (int off = 128; off > 0; off >>= 1) {
        if (tid < off) { s1[tid] += s1[tid + off]; s2[tid] += s2[tid + off]; }
        __syncthreads();
    }
    const float mean = s1[0] / (float)C;
    const float var  = s2[0] / (float)C - mean * mean;
    const float rs   = rsqrtf(var + 1e-5f);
    y[(size_t)row * C + tid]       = f2bf((v0 - mean) * rs * g[tid]       + b[tid]);
    y[(size_t)row * C + tid + 256] = f2bf((v1 - mean) * rs * g[tid + 256] + b[tid + 256]);
}

// ---------------------------------------------------------------------------
// Causal linear attention scan, phi(x)=elu(x)+1; fp32 qkv in, bf16 out.
// One 64-thread block per (batch, head); thread e owns KV-state column e.
// ---------------------------------------------------------------------------
__device__ __forceinline__ float phi_fn(float x) { return x > 0.0f ? x + 1.0f : __expf(x); }

__global__ __launch_bounds__(64) void linattn_kernel(
    const float* __restrict__ qkv, unsigned short* __restrict__ out, int T, int C)
{
    const int b  = blockIdx.x >> 3;   // H = 8
    const int hh = blockIdx.x & 7;
    const int e  = threadIdx.x;       // 0..63 (D = 64)
    __shared__ float Qs[64], Ks[64], Kc[64];

    float S[64];
#pragma unroll
    for (int d = 0; d < 64; ++d) S[d] = 0.0f;
    Kc[e] = 0.0f;
    __syncthreads();

    for (int t = 0; t < T; ++t) {
        const size_t base = (size_t)(b * T + t) * 3 * C;
        const float qv = qkv[base +         hh * 64 + e];
        const float kv = qkv[base + C +     hh * 64 + e];
        const float ve = qkv[base + 2 * C + hh * 64 + e];
        Qs[e] = phi_fn(qv);
        const float ke = phi_fn(kv);
        Ks[e] = ke;
        Kc[e] += ke;
        __syncthreads();

        float num = 0.0f, den = 0.0f;
#pragma unroll
        for (int d = 0; d < 64; ++d) {
            const float qd = Qs[d];
            S[d] += Ks[d] * ve;
            num  += qd * S[d];
            den  += qd * Kc[d];
        }
        out[(size_t)(b * T + t) * C + hh * 64 + e] = f2bf(num / (den + 1e-6f));
        __syncthreads();
    }
}

// ---------------------------------------------------------------------------
// values head: out[row] = h[row,:] @ w[:,0] + b   (h is bf16)
// ---------------------------------------------------------------------------
__global__ __launch_bounds__(256) void values_kernel(
    const unsigned short* __restrict__ h, const float* __restrict__ w,
    const float* __restrict__ bia, float* __restrict__ out, int M, int C)
{
    const int row = blockIdx.x * blockDim.x + threadIdx.x;
    if (row >= M) return;
    float acc = bia[0];
    const unsigned short* hr = h + (size_t)row * C;
    for (int c = 0; c < C; ++c) acc += bf2f(hr[c]) * w[c];
    out[row] = acc;
}

// ---------------------------------------------------------------------------
extern "C" void kernel_launch(void* const* d_in, const int* in_sizes, int n_in,
                              void* d_out, int out_size, void* d_ws, size_t ws_size,
                              hipStream_t stream)
{
    (void)in_sizes; (void)n_in; (void)out_size; (void)ws_size;

    const float* states = (const float*)d_in[0];
    const float* sew    = (const float*)d_in[1];
    const float* seb    = (const float*)d_in[2];
    const float* pos    = (const float*)d_in[3];
    const float* ln1_g  = (const float*)d_in[4];
    const float* ln1_b  = (const float*)d_in[5];
    const float* qkv_w  = (const float*)d_in[6];
    const float* qkv_b  = (const float*)d_in[7];
    const float* ow     = (const float*)d_in[8];
    const float* ob     = (const float*)d_in[9];
    const float* ln2_g  = (const float*)d_in[10];
    const float* ln2_b  = (const float*)d_in[11];
    const float* w1     = (const float*)d_in[12];
    const float* b1     = (const float*)d_in[13];
    const float* w2     = (const float*)d_in[14];
    const float* b2     = (const float*)d_in[15];
    const float* lnf_g  = (const float*)d_in[16];
    const float* lnf_b  = (const float*)d_in[17];
    const float* amw    = (const float*)d_in[18];
    const float* amb    = (const float*)d_in[19];
    const float* alsw   = (const float*)d_in[20];
    const float* alsb   = (const float*)d_in[21];
    const float* crw    = (const float*)d_in[22];
    const float* crb    = (const float*)d_in[23];

    const int T = 1024, M = 2048, C = 512, SS = 512, F = 2048, L = 4, A = 64;

    // ---- workspace: fp32 region then bf16 region (all 16B aligned) ----
    float* fp = (float*)d_ws;
    float* x   = fp; fp += (size_t)M * C;          // residual stream
    float* qkv = fp; fp += (size_t)M * 3 * C;      // qkv activations
    unsigned short* us = (unsigned short*)fp;
    unsigned short* h_bf   = us; us += (size_t)M * C;
    unsigned short* attn_bf= us; us += (size_t)M * C;
    unsigned short* mlp_bf = us; us += (size_t)M * F;
    unsigned short* st_bf  = us; us += (size_t)M * SS;
    unsigned short* wt_sew = us; us += (size_t)C * SS;
    unsigned short* wt_qkv = us; us += (size_t)L * 3 * C * C;
    unsigned short* wt_ow  = us; us += (size_t)L * C * C;
    unsigned short* wt_w1  = us; us += (size_t)L * F * C;
    unsigned short* wt_w2  = us; us += (size_t)L * C * F;
    unsigned short* wt_am  = us; us += (size_t)A * C;
    unsigned short* wt_als = us; us += (size_t)A * C;

    const dim3 blk(256);
    auto ggrid = [&](int N) { return dim3((unsigned)((N + 127) / 128), (unsigned)(M / 64)); };
    auto tgrid = [](int K, int N) { return dim3((unsigned)((N + 31) / 32), (unsigned)((K + 31) / 32)); };

    // ---- one-time (per launch) weight transpose+convert, activation convert ----
    convert_bf16_kernel<<<512, 256, 0, stream>>>(states, st_bf, M * SS);
    transpose_bf16_kernel<<<tgrid(SS, C), blk, 0, stream>>>(sew, wt_sew, SS, C);
    for (int i = 0; i < L; ++i) {
        transpose_bf16_kernel<<<tgrid(C, 3 * C), blk, 0, stream>>>(
            qkv_w + (size_t)i * C * 3 * C, wt_qkv + (size_t)i * 3 * C * C, C, 3 * C);
        transpose_bf16_kernel<<<tgrid(C, C), blk, 0, stream>>>(
            ow + (size_t)i * C * C, wt_ow + (size_t)i * C * C, C, C);
        transpose_bf16_kernel<<<tgrid(C, F), blk, 0, stream>>>(
            w1 + (size_t)i * C * F, wt_w1 + (size_t)i * F * C, C, F);
        transpose_bf16_kernel<<<tgrid(F, C), blk, 0, stream>>>(
            w2 + (size_t)i * F * C, wt_w2 + (size_t)i * C * F, F, C);
    }
    transpose_bf16_kernel<<<tgrid(C, A), blk, 0, stream>>>(amw, wt_am, C, A);
    transpose_bf16_kernel<<<tgrid(C, A), blk, 0, stream>>>(alsw, wt_als, C, A);

    // ---- forward pass ----
    // x = states @ sew + seb + pos[t % 1024]
    gemm_wmma_kernel<ACT_NONE, true, false><<<ggrid(C), blk, 0, stream>>>(
        st_bf, wt_sew, seb, pos, 1024, x, M, C, SS);

    for (int i = 0; i < L; ++i) {
        ln_kernel<<<M, 256, 0, stream>>>(x, ln1_g + i * C, ln1_b + i * C, h_bf, C);
        gemm_wmma_kernel<ACT_NONE, false, false><<<ggrid(3 * C), blk, 0, stream>>>(
            h_bf, wt_qkv + (size_t)i * 3 * C * C, qkv_b + (size_t)i * 3 * C,
            nullptr, 0, qkv, M, 3 * C, C);
        linattn_kernel<<<16, 64, 0, stream>>>(qkv, attn_bf, T, C);
        gemm_wmma_kernel<ACT_NONE, true, false><<<ggrid(C), blk, 0, stream>>>(
            attn_bf, wt_ow + (size_t)i * C * C, ob + (size_t)i * C, x, 0, x, M, C, C);
        ln_kernel<<<M, 256, 0, stream>>>(x, ln2_g + i * C, ln2_b + i * C, h_bf, C);
        gemm_wmma_kernel<ACT_GELU, false, true><<<ggrid(F), blk, 0, stream>>>(
            h_bf, wt_w1 + (size_t)i * F * C, b1 + (size_t)i * F, nullptr, 0, mlp_bf, M, F, C);
        gemm_wmma_kernel<ACT_NONE, true, false><<<ggrid(C), blk, 0, stream>>>(
            mlp_bf, wt_w2 + (size_t)i * C * F, b2 + (size_t)i * C, x, 0, x, M, C, F);
    }

    ln_kernel<<<M, 256, 0, stream>>>(x, lnf_g, lnf_b, h_bf, C);

    float* logits = (float*)d_out;
    float* logstd = logits + (size_t)M * A;
    float* values = logstd + (size_t)M * A;
    gemm_wmma_kernel<ACT_NONE, false, false><<<ggrid(A), blk, 0, stream>>>(
        h_bf, wt_am, amb, nullptr, 0, logits, M, A, C);
    gemm_wmma_kernel<ACT_CLIP, false, false><<<ggrid(A), blk, 0, stream>>>(
        h_bf, wt_als, alsb, nullptr, 0, logstd, M, A, C);
    values_kernel<<<(M + 255) / 256, 256, 0, stream>>>(h_bf, crw, crb, values, M, C);
}
